// Class_2_6305011990839
// MI455X (gfx1250) — matrix operand order repaired and checked by reading, not compiled, not run on hardware
//
#include <hip/hip_runtime.h>
#include <math.h>

// ============================================================================
// MI455X (gfx1250) MulT-style multimodal transformer forward.
// ~1.7 TFLOP/forward, matrix-op bound => V_WMMA_F32_16X16X32_BF16 everywhere
// (bf16 inputs, fp32 accumulate). fp32 weights convert to bf16 while staging
// into LDS (all weight matrices fit the 192MB L2, so M-tile reuse is free).
// Interior tiles are staged with CDNA5 async HBM->LDS copies
// (global_load_async_to_lds_b128) into double-buffered LDS, prefetched one
// K-tile ahead and synchronized with partial s_wait_asynccnt waits.
// ============================================================================

#define EPS_LN 1e-6f
#ifndef USE_ASYNC
#define USE_ASYNC 1
#endif

typedef __bf16 bf16;
typedef __attribute__((ext_vector_type(16))) __bf16 v16bf;
typedef __attribute__((ext_vector_type(8)))  float  v8f;

__device__ __forceinline__ bf16 bf_zero() {
  return __builtin_bit_cast(bf16, (unsigned short)0);
}
__device__ __forceinline__ bf16 f2b(float f) {
  unsigned u = __float_as_uint(f);
  u += 0x7fffu + ((u >> 16) & 1u);           // round-to-nearest-even
  return __builtin_bit_cast(bf16, (unsigned short)(u >> 16));
}
__device__ __forceinline__ unsigned short f2b_bits(float f) {
  unsigned u = __float_as_uint(f);
  u += 0x7fffu + ((u >> 16) & 1u);
  return (unsigned short)(u >> 16);
}
__device__ __forceinline__ bf16 to_bf(float x) { return f2b(x); }
__device__ __forceinline__ bf16 to_bf(bf16 x)  { return x; }
__device__ __forceinline__ unsigned short to_bf_bits(float x) { return f2b_bits(x); }
__device__ __forceinline__ unsigned short to_bf_bits(bf16 x)  {
  return __builtin_bit_cast(unsigned short, x);
}
__device__ __forceinline__ int imin(int a, int b) { return a < b ? a : b; }

// LDS byte offset of a __shared__ object (flat LDS address low 32 bits)
__device__ __forceinline__ unsigned lds_off(const void* p) {
  return (unsigned)(unsigned long long)p;
}
// CDNA5 async HBM -> LDS 16-byte copy (VGLOBAL GLOBAL_LOAD_ASYNC_TO_LDS_B128)
__device__ __forceinline__ void async_b128(unsigned lds_addr, const void* gaddr) {
  asm volatile("global_load_async_to_lds_b128 %0, %1, off"
               :: "v"(lds_addr), "v"(gaddr) : "memory");
}

// ----------------------------------------------------------------------------
// WMMA GEMM: C[z] = act(A[z] @ op(B[z]) + bias + res)
//   A: bf16 MxK row-major (lda); B: BT (fp32 weights or bf16 activations);
//   TRANSB => B is NxK row-major. Optional fp32 Cf and/or bf16 Cb outputs.
//   res: optional fp32 residual; resRowMod!=0 => residual row = row % mod
//   (broadcast position embeddings). Batch z = z1*zsplit + z0 with two-term
//   strides per operand (head-strided attention batching without reshapes).
// Macro tile 128x64, BK=32, 256 threads = 8 wave32 waves (4m x 2n); each wave
// owns 32x32 = four 16x16 WMMA accumulators. Double-buffered LDS, async
// prefetch one K-tile ahead, partial ASYNCcnt waits.
// ----------------------------------------------------------------------------
template <typename BT, bool TRANSB, int ACT>
__global__ __launch_bounds__(256) void wmma_gemm(
    const bf16* __restrict__ A, long long sA1, long long sA0, int lda,
    const BT*  __restrict__ Bm, long long sB1, long long sB0, int ldb,
    const float* __restrict__ bias,
    const float* __restrict__ res, int ldr, int resRowMod,
    float* __restrict__ Cf, long long sC1, long long sC0, int ldc,
    bf16*  __restrict__ Cb, long long sCb1, long long sCb0, int ldcb,
    int M, int N, int K, int zsplit)
{
  __shared__ bf16 As[2][128][40];  // [buf][m][k], 80B rows keep 16B alignment
  __shared__ bf16 Bs[2][64][40];   // [buf][n][k]: (k,k+1) pairs are one b32

  const int tid = threadIdx.x;
  const int z  = blockIdx.z;
  const int z1 = z / zsplit, z0 = z - z1 * zsplit;
  A  += z1 * sA1 + z0 * sA0;
  Bm += z1 * sB1 + z0 * sB0;
  if (Cf) Cf += z1 * sC1  + z0 * sC0;
  if (Cb) Cb += z1 * sCb1 + z0 * sCb0;

  const int tileM = blockIdx.y * 128;
  const int tileN = blockIdx.x * 64;

  const int wave = tid >> 5;        // wave32
  const int lane = tid & 31;
  const int l15  = lane & 15;
  const int hi   = lane >> 4;       // 0 | 1
  const int wm   = wave & 3;        // 4 m-slabs of 32 rows
  const int wn   = wave >> 2;       // 2 n-slabs of 32 cols

  // ISA §7.12.2 fragment layouts
  const int arow0 = wm * 32 + l15;  // A frag rows: +0 and +16
  const int kbA   = hi ? 8 : 0;
  const int kbB   = hi ? 16 : 0;
  const int bcol0 = wn * 32 + l15;  // B frag cols: +0 and +16

  const v8f z8 = {0.f,0.f,0.f,0.f,0.f,0.f,0.f,0.f};
  v8f acc[2][2] = {{z8, z8}, {z8, z8}};

  // staging roles
  const int aRow = tid >> 1;        // 0..127
  const int aK   = (tid & 1) * 16;  // 16 bf16 = 32B per thread
  const int bN   = tid >> 2;        // 0..63
  const int bK   = (tid & 3) * 8;   // 8 elements per thread

  const bool fullMN = (tileM + 128 <= M) && (tileN + 64 <= N);
  const int  ntiles = (K + 31) / 32;

  union Frag { v16bf v; unsigned u[8]; };

  // Stage one 128x32 A tile + 64x32 B tile into LDS buffer `buf`.
  auto stage = [&](int k0, int buf) {
    if (fullMN && (k0 + 32 <= K)) {
      // ---- interior: async / vector staging, no guards ----
      const bf16* ga = A + (long long)(tileM + aRow) * lda + (k0 + aK);
      const unsigned la = lds_off(&As[buf][aRow][aK]);
#if USE_ASYNC
      async_b128(la,      ga);
      async_b128(la + 16, ga + 8);
#else
      *reinterpret_cast<uint4*>(&As[buf][aRow][aK])     = *reinterpret_cast<const uint4*>(ga);
      *reinterpret_cast<uint4*>(&As[buf][aRow][aK + 8]) = *reinterpret_cast<const uint4*>(ga + 8);
#endif
      if constexpr (TRANSB) {
        // bf16 activations, N x K row-major: contiguous 16B per thread
        const BT* gb = Bm + (long long)(tileN + bN) * ldb + (k0 + bK);
        const unsigned lb = lds_off(&Bs[buf][bN][bK]);
#if USE_ASYNC
        async_b128(lb, gb);
#else
        *reinterpret_cast<uint4*>(&Bs[buf][bN][bK]) = *reinterpret_cast<const uint4*>(gb);
#endif
      } else {
        // K x N source: gather 8 k's for one n, convert, one ds_store_b128
        const int gn = tileN + bN;
        unsigned short h[8];
        #pragma unroll
        for (int j = 0; j < 8; ++j)
          h[j] = to_bf_bits(Bm[(long long)(k0 + bK + j) * ldb + gn]);
        uint4 pk;
        pk.x = (unsigned)h[0] | ((unsigned)h[1] << 16);
        pk.y = (unsigned)h[2] | ((unsigned)h[3] << 16);
        pk.z = (unsigned)h[4] | ((unsigned)h[5] << 16);
        pk.w = (unsigned)h[6] | ((unsigned)h[7] << 16);
        *reinterpret_cast<uint4*>(&Bs[buf][bN][bK]) = pk;
      }
    } else {
      // ---- edge tile: branch-free clamped staging (no exec-mask dance) ----
      {
        const int gr  = tileM + aRow;
        const int grc = imin(gr, M - 1);
        #pragma unroll
        for (int j = 0; j < 16; ++j) {
          const int gk  = k0 + aK + j;
          const int gkc = imin(gk, K - 1);
          const bf16 v = A[(long long)grc * lda + gkc];
          As[buf][aRow][aK + j] = (gr < M && gk < K) ? v : bf_zero();
        }
      }
      {
        const int gn  = tileN + bN;
        const int gnc = imin(gn, N - 1);
        #pragma unroll
        for (int j = 0; j < 8; ++j) {
          const int gk  = k0 + bK + j;
          const int gkc = imin(gk, K - 1);
          const long long idx = TRANSB ? ((long long)gnc * ldb + gkc)
                                       : ((long long)gkc * ldb + gnc);
          const bf16 v = to_bf(Bm[idx]);
          Bs[buf][bN][bK + j] = (gn < N && gk < K) ? v : bf_zero();
        }
      }
    }
  };

  stage(0, 0);                       // prologue: tile 0 into buffer 0
  for (int t = 0; t < ntiles; ++t) {
    const int buf = t & 1;
    const bool haveNext = (t + 1) < ntiles;
    if (haveNext) stage((t + 1) * 32, buf ^ 1);     // prefetch one tile ahead
#if USE_ASYNC
    // Partial wait: only the OLDER async batch (tile t) must have landed; the
    // batch just issued for tile t+1 (3 asyncs if TRANSB, else 2) may still
    // be in flight. If tile t+1 was edge-staged (no asyncs), drain fully.
    const bool relaxed = haveNext && fullMN && ((t + 1) * 32 + 32 <= K);
    if (relaxed) {
      if constexpr (TRANSB) asm volatile("s_wait_asynccnt 0x3" ::: "memory");
      else                  asm volatile("s_wait_asynccnt 0x2" ::: "memory");
    } else {
      asm volatile("s_wait_asynccnt 0x0" ::: "memory");
    }
#endif
    __syncthreads();                 // tile t visible to all waves

    Frag fa[2], fb[2];
    #pragma unroll
    for (int i = 0; i < 8; ++i) {
      const int ka = ((i & 4) ? 16 : 0) + kbA + 2 * (i & 3);
      fa[0].u[i] = *reinterpret_cast<const unsigned*>(&As[buf][arow0][ka]);
      fa[1].u[i] = *reinterpret_cast<const unsigned*>(&As[buf][arow0 + 16][ka]);
      const int kb = ka;
      fb[0].u[i] = *reinterpret_cast<const unsigned*>(&Bs[buf][bcol0][kb]);
      fb[1].u[i] = *reinterpret_cast<const unsigned*>(&Bs[buf][bcol0 + 16][kb]);
    }
    #pragma unroll
    for (int mi = 0; mi < 2; ++mi)
      #pragma unroll
      for (int ni = 0; ni < 2; ++ni)
        acc[mi][ni] = __builtin_amdgcn_wmma_f32_16x16x32_bf16(
            false, fa[mi].v, false, fb[ni].v, (short)0, acc[mi][ni], false, false);
    __syncthreads();                 // reads done before buf is re-staged
  }

  // Epilogue: C/D layout — VGPR r holds M = r + 8*hi, N = l15
  #pragma unroll
  for (int mi = 0; mi < 2; ++mi) {
    #pragma unroll
    for (int ni = 0; ni < 2; ++ni) {
      const int col = tileN + wn * 32 + ni * 16 + l15;
      if (col >= N) continue;
      #pragma unroll
      for (int r = 0; r < 8; ++r) {
        const int row = tileM + wm * 32 + mi * 16 + r + hi * 8;
        if (row >= M) continue;
        float v = acc[mi][ni][r];
        if (bias) v += bias[col];
        if (res) {
          const int rr = resRowMod ? (row % resRowMod) : row;
          v += res[(long long)rr * ldr + col];
        }
        if (ACT == 1) v = v > 0.f ? v : 0.f;
        if (Cf) Cf[(long long)row * ldc  + col] = v;
        if (Cb) Cb[(long long)row * ldcb + col] = f2b(v);
      }
    }
  }
}

// ----------------------------------------------------------------------------
// LayerNorm over rows of length D (fp32 in, bf16 out, optional fused ReLU)
// ----------------------------------------------------------------------------
__global__ void layernorm_kernel(const float* __restrict__ X, int ldx,
                                 const float* __restrict__ g,
                                 const float* __restrict__ b,
                                 bf16* __restrict__ Y, int ldy,
                                 int D, float eps, int relu)
{
  __shared__ float s1[256], s2[256];
  const int row = blockIdx.x, tid = threadIdx.x;
  const float* x = X + (long long)row * ldx;
  float a1 = 0.f, a2 = 0.f;
  for (int c = tid; c < D; c += 256) { float v = x[c]; a1 += v; a2 += v * v; }
  s1[tid] = a1; s2[tid] = a2; __syncthreads();
  for (int s = 128; s > 0; s >>= 1) {
    if (tid < s) { s1[tid] += s1[tid + s]; s2[tid] += s2[tid + s]; }
    __syncthreads();
  }
  const float mean = s1[0] / (float)D;
  const float var  = s2[0] / (float)D - mean * mean;
  const float rstd = rsqrtf(var + eps);
  bf16* y = Y + (long long)row * ldy;
  for (int c = tid; c < D; c += 256) {
    float v = (x[c] - mean) * rstd * g[c] + b[c];
    if (relu) v = v > 0.f ? v : 0.f;
    y[c] = f2b(v);
  }
}

// ----------------------------------------------------------------------------
// Masked softmax rows (fused 1/sqrt(dh) scale), fp32 scores -> bf16 probs
// row = (b*H + h)*Sq + q ; mask is (B, Sk)
// ----------------------------------------------------------------------------
__global__ void softmax_kernel(const float* __restrict__ S,
                               const float* __restrict__ mask,
                               bf16* __restrict__ P,
                               int Sk, int HSq, float scale)
{
  __shared__ float sh[256];
  const int row = blockIdx.x, tid = threadIdx.x;
  const int bidx = row / HSq;
  const float* s = S + (long long)row * Sk;
  const float* m = mask + (long long)bidx * Sk;

  float mx = -3.0e38f;
  for (int c = tid; c < Sk; c += 256) {
    float v = s[c] * scale - 10000.f * (1.f - m[c]);
    mx = fmaxf(mx, v);
  }
  sh[tid] = mx; __syncthreads();
  for (int st = 128; st > 0; st >>= 1) {
    if (tid < st) sh[tid] = fmaxf(sh[tid], sh[tid + st]);
    __syncthreads();
  }
  const float rowmax = sh[0]; __syncthreads();

  float sum = 0.f;
  for (int c = tid; c < Sk; c += 256)
    sum += __expf(s[c] * scale - 10000.f * (1.f - m[c]) - rowmax);
  sh[tid] = sum; __syncthreads();
  for (int st = 128; st > 0; st >>= 1) {
    if (tid < st) sh[tid] += sh[tid + st];
    __syncthreads();
  }
  const float inv = 1.f / sh[0];

  bf16* p = P + (long long)row * Sk;
  for (int c = tid; c < Sk; c += 256) {
    float v = __expf(s[c] * scale - 10000.f * (1.f - m[c]) - rowmax) * inv;
    p[c] = f2b(v);
  }
}

// fp32 (rows x K) -> bf16 (rows x ldd), ldd >= K padded for 16B row alignment
__global__ void cvt_pad_bf16(const float* __restrict__ x, bf16* __restrict__ y,
                             int rows, int K, int ldd)
{
  const long long i = (long long)blockIdx.x * blockDim.x + threadIdx.x;
  if (i >= (long long)rows * K) return;
  const int r = (int)(i / K), c = (int)(i - (long long)r * K);
  y[(long long)r * ldd + c] = f2b(x[i]);
}

// mean/max pooling over concat([lc, ac, vc], axis=1): each (B,S,D3)
// out (B, 2*D3) bf16: [0,D3)=mean, [D3,2*D3)=max  (order-invariant reductions)
__global__ void pool_kernel(const float* __restrict__ lc,
                            const float* __restrict__ ac,
                            const float* __restrict__ vc,
                            int B, int S, int D3, bf16* __restrict__ out)
{
  const int i = blockIdx.x * blockDim.x + threadIdx.x;
  if (i >= B * D3) return;
  const int b = i / D3, c = i - b * D3;
  float sum = 0.f, mx = -3.0e38f;
  const float* ptrs[3] = { lc, ac, vc };
  for (int p = 0; p < 3; ++p) {
    const float* base = ptrs[p] + (long long)b * S * D3 + c;
    for (int s = 0; s < S; ++s) {
      float v = base[(long long)s * D3];
      sum += v; mx = fmaxf(mx, v);
    }
  }
  out[(long long)b * (2 * D3) + c]      = f2b(sum / (3.f * (float)S));
  out[(long long)b * (2 * D3) + D3 + c] = f2b(mx);
}

// ============================================================================
// Host-side orchestration
// ============================================================================
struct BlockW {
  const float *ln_g, *ln_b;
  const float *wq, *bq, *wk, *bk, *wv, *bv, *wo, *bo, *w1, *b1, *w2, *b2;
};
struct Scratch {
  bf16 *qn, *kn, *vn, *qh, *kh, *vh, *attn, *hh, *P, *f1;
  float *scores, *x1;
};

static void gemm_w(hipStream_t st, const bf16* A, int lda,
                   const float* W, int ldw,
                   const float* bias, const float* res, int ldr, int rmod,
                   float* Cf, int ldc, bf16* Cb, int ldcb,
                   int M, int N, int K, bool relu)
{
  dim3 g((N + 63) / 64, (M + 127) / 128, 1);
  if (relu)
    wmma_gemm<float, false, 1><<<g, 256, 0, st>>>(
        A, 0, 0, lda, W, 0, 0, ldw, bias, res, ldr, rmod,
        Cf, 0, 0, ldc, Cb, 0, 0, ldcb, M, N, K, 1);
  else
    wmma_gemm<float, false, 0><<<g, 256, 0, st>>>(
        A, 0, 0, lda, W, 0, 0, ldw, bias, res, ldr, rmod,
        Cf, 0, 0, ldc, Cb, 0, 0, ldcb, M, N, K, 1);
}

// One transformer block: pre-LN q/k/v, MHSA + out-proj residual, LN + ReLU FFN
// residual. qout may alias qin (FFN2 epilogue reads only x1, writes qout once).
static void run_block(hipStream_t st, const BlockW& w,
                      const float* qin, int ldq,
                      const float* kvin, int ldkv,
                      const float* mask,
                      float* qout, int ldqout,
                      int D, int DFF, const Scratch& sc)
{
  const int B = 16, S = 256, H = 8, M = B * S;
  const int dh = D / H;
  const float scale = 1.0f / sqrtf((float)dh);

  layernorm_kernel<<<M, 256, 0, st>>>(qin,  ldq,  w.ln_g + 0 * D, w.ln_b + 0 * D, sc.qn, D, D, EPS_LN, 0);
  layernorm_kernel<<<M, 256, 0, st>>>(kvin, ldkv, w.ln_g + 1 * D, w.ln_b + 1 * D, sc.kn, D, D, EPS_LN, 0);
  layernorm_kernel<<<M, 256, 0, st>>>(kvin, ldkv, w.ln_g + 2 * D, w.ln_b + 2 * D, sc.vn, D, D, EPS_LN, 0);

  gemm_w(st, sc.qn, D, w.wq, D, w.bq, nullptr, 0, 0, nullptr, 0, sc.qh, D, M, D, D, false);
  gemm_w(st, sc.kn, D, w.wk, D, w.bk, nullptr, 0, 0, nullptr, 0, sc.kh, D, M, D, D, false);
  gemm_w(st, sc.vn, D, w.wv, D, w.bv, nullptr, 0, 0, nullptr, 0, sc.vh, D, M, D, D, false);

  {   // scores[b,h] = Qh @ Kh^T (head-strided batch z = b*H + h), async staging
    dim3 g((S + 63) / 64, (S + 127) / 128, B * H);
    wmma_gemm<bf16, true, 0><<<g, 256, 0, st>>>(
        sc.qh, (long long)S * D, dh, D,
        sc.kh, (long long)S * D, dh, D,
        nullptr, nullptr, 0, 0,
        sc.scores, (long long)H * S * S, (long long)S * S, S,
        nullptr, 0, 0, 0,
        S, S, dh, H);
  }
  softmax_kernel<<<B * H * S, 256, 0, st>>>(sc.scores, mask, sc.P, S, H * S, scale);
  {   // attn[b,h] = P @ Vh, written head-strided straight into (B,S,D)
    dim3 g((dh + 63) / 64, (S + 127) / 128, B * H);
    wmma_gemm<bf16, false, 0><<<g, 256, 0, st>>>(
        sc.P, (long long)H * S * S, (long long)S * S, S,
        sc.vh, (long long)S * D, dh, D,
        nullptr, nullptr, 0, 0,
        nullptr, 0, 0, 0,
        sc.attn, (long long)S * D, dh, D,
        S, dh, S, H);
  }
  // x1 = qin + attn @ wo + bo
  gemm_w(st, sc.attn, D, w.wo, D, w.bo, qin, ldq, 0, sc.x1, D, nullptr, 0, M, D, D, false);
  // hh = LN(x1)
  layernorm_kernel<<<M, 256, 0, st>>>(sc.x1, D, w.ln_g + 3 * D, w.ln_b + 3 * D, sc.hh, D, D, EPS_LN, 0);
  // f1 = relu(hh @ w1 + b1)
  gemm_w(st, sc.hh, D, w.w1, DFF, w.b1, nullptr, 0, 0, nullptr, 0, sc.f1, DFF, M, DFF, D, true);
  // qout = x1 + f1 @ w2 + b2
  gemm_w(st, sc.f1, DFF, w.w2, D, w.b2, sc.x1, D, 0, qout, ldqout, nullptr, 0, M, D, DFF, false);
}

// Input index map (setup_inputs insertion order, params flattened depth-first)
enum {
  IN_L = 0, IN_V, IN_A, IN_LM, IN_VM, IN_AM,
  P_CONV_L, P_CONV_V, P_CONV_A, P_POS_L, P_POS_V, P_POS_A,
  MM_LN_G, MM_LN_B, MM_WQ, MM_BQ, MM_WK, MM_BK, MM_WV, MM_BV,
  MM_WO, MM_BO, MM_W1, MM_B1, MM_W2, MM_B2,
  FU_LN_G, FU_LN_B, FU_WQ, FU_BQ, FU_WK, FU_BK, FU_WV, FU_BV,
  FU_WO, FU_BO, FU_W1, FU_B1, FU_W2, FU_B2,
  P_FC_W, P_FC_B, P_LN_G, P_LN_B, P_CLS_W, P_CLS_B
};

extern "C" void kernel_launch(void* const* d_in, const int* in_sizes, int n_in,
                              void* d_out, int out_size, void* d_ws, size_t ws_size,
                              hipStream_t stream)
{
  (void)in_sizes; (void)n_in; (void)out_size; (void)ws_size;

  const int B = 16, S = 256, M = B * S;
  const int D0 = 512, D3 = 1536, DFF0 = 2048, DFF3 = 6144;
  const int KL = 300, KV = 35, KA = 74;
  const int KLp = 304, KVp = 48, KAp = 80;   // padded ld for 16B-aligned rows

  const float* l_in  = (const float*)d_in[IN_L];
  const float* v_in  = (const float*)d_in[IN_V];
  const float* a_in  = (const float*)d_in[IN_A];
  const float* lmask = (const float*)d_in[IN_LM];
  const float* vmask = (const float*)d_in[IN_VM];
  const float* amask = (const float*)d_in[IN_AM];
  const float* conv_l = (const float*)d_in[P_CONV_L];
  const float* conv_v = (const float*)d_in[P_CONV_V];
  const float* conv_a = (const float*)d_in[P_CONV_A];
  const float* pos_l  = (const float*)d_in[P_POS_L];
  const float* pos_v  = (const float*)d_in[P_POS_V];
  const float* pos_a  = (const float*)d_in[P_POS_A];
  const float* mm_p[14]; for (int i = 0; i < 14; ++i) mm_p[i] = (const float*)d_in[MM_LN_G + i];
  const float* fu_p[14]; for (int i = 0; i < 14; ++i) fu_p[i] = (const float*)d_in[FU_LN_G + i];
  const float* fc_w  = (const float*)d_in[P_FC_W];
  const float* fc_b  = (const float*)d_in[P_FC_B];
  const float* ln_g  = (const float*)d_in[P_LN_G];
  const float* ln_b  = (const float*)d_in[P_LN_B];
  const float* cls_w = (const float*)d_in[P_CLS_W];
  const float* cls_b = (const float*)d_in[P_CLS_B];
  float* outp = (float*)d_out;

  // ---- workspace bump allocator --------------------------------------------
  char* base = (char*)d_ws;
  size_t off = 0;
  auto alloc = [&](size_t bytes) -> void* {
    void* p = base + off;
    off += (bytes + 255) & ~(size_t)255;
    return p;
  };
  bf16* l_bf = (bf16*)alloc((size_t)M * KLp * 2);
  bf16* v_bf = (bf16*)alloc((size_t)M * KVp * 2);
  bf16* a_bf = (bf16*)alloc((size_t)M * KAp * 2);
  float* lbuf = (float*)alloc((size_t)M * D0 * 4);
  float* vbuf = (float*)alloc((size_t)M * D0 * 4);
  float* abuf = (float*)alloc((size_t)M * D0 * 4);
  float* lc  = (float*)alloc((size_t)M * D3 * 4);
  float* vc3 = (float*)alloc((size_t)M * D3 * 4);
  float* ac3 = (float*)alloc((size_t)M * D3 * 4);

  Scratch sc;
  sc.qn   = (bf16*)alloc((size_t)M * D3 * 2);
  sc.kn   = (bf16*)alloc((size_t)M * D3 * 2);
  sc.vn   = (bf16*)alloc((size_t)M * D3 * 2);
  sc.qh   = (bf16*)alloc((size_t)M * D3 * 2);
  sc.kh   = (bf16*)alloc((size_t)M * D3 * 2);
  sc.vh   = (bf16*)alloc((size_t)M * D3 * 2);
  sc.attn = (bf16*)alloc((size_t)M * D3 * 2);
  sc.hh   = (bf16*)alloc((size_t)M * D3 * 2);
  sc.scores = (float*)alloc((size_t)B * 8 * S * S * 4);
  sc.P    = (bf16*)alloc((size_t)B * 8 * S * S * 2);
  sc.x1   = (float*)alloc((size_t)M * D3 * 4);
  sc.f1   = (bf16*)alloc((size_t)M * DFF3 * 2);
  bf16*  pooled = (bf16*)alloc((size_t)B * 2 * D3 * 2);
  float* fcout  = (float*)alloc((size_t)B * D0 * 4);
  bf16*  fcln   = (bf16*)alloc((size_t)B * D0 * 2);

  // ---- input convert (padded ld) + conv1d(k=1) + position embeddings -------
  cvt_pad_bf16<<<(M * KL + 255) / 256, 256, 0, stream>>>(l_in, l_bf, M, KL, KLp);
  cvt_pad_bf16<<<(M * KV + 255) / 256, 256, 0, stream>>>(v_in, v_bf, M, KV, KVp);
  cvt_pad_bf16<<<(M * KA + 255) / 256, 256, 0, stream>>>(a_in, a_bf, M, KA, KAp);
  gemm_w(stream, l_bf, KLp, conv_l, D0, nullptr, pos_l, D0, S, lbuf, D0, nullptr, 0, M, D0, KL, false);
  gemm_w(stream, v_bf, KVp, conv_v, D0, nullptr, pos_v, D0, S, vbuf, D0, nullptr, 0, M, D0, KV, false);
  gemm_w(stream, a_bf, KAp, conv_a, D0, nullptr, pos_a, D0, S, abuf, D0, nullptr, 0, M, D0, KA, false);

  // ---- cross-modal blocks (only layer i=L-1 survives in the reference) -----
  auto mmW = [&](int bi) -> BlockW {
    BlockW w; const long long o = (long long)bi * 2 + 1;  // (bi, L-1)
    w.ln_g = mm_p[0] + o * 4 * D0;  w.ln_b = mm_p[1] + o * 4 * D0;
    w.wq = mm_p[2] + o * D0 * D0;   w.bq = mm_p[3] + o * D0;
    w.wk = mm_p[4] + o * D0 * D0;   w.bk = mm_p[5] + o * D0;
    w.wv = mm_p[6] + o * D0 * D0;   w.bv = mm_p[7] + o * D0;
    w.wo = mm_p[8] + o * D0 * D0;   w.bo = mm_p[9] + o * D0;
    w.w1 = mm_p[10] + o * D0 * DFF0; w.b1 = mm_p[11] + o * DFF0;
    w.w2 = mm_p[12] + o * DFF0 * D0; w.b2 = mm_p[13] + o * D0;
    return w;
  };
  struct Cfg { const float* q; const float* kv; const float* mask; float* out; };
  const Cfg mmcfg[9] = {
    { lbuf, lbuf, lmask, lc  + 0    },   // ll
    { lbuf, vbuf, vmask, lc  + 512  },   // lv
    { lbuf, abuf, amask, lc  + 1024 },   // la
    { vbuf, vbuf, vmask, vc3 + 0    },   // vv
    { vbuf, lbuf, lmask, vc3 + 512  },   // vl
    { vbuf, abuf, amask, vc3 + 1024 },   // va
    { abuf, abuf, amask, ac3 + 0    },   // aa
    { abuf, lbuf, lmask, ac3 + 512  },   // al
    { abuf, vbuf, vmask, ac3 + 1024 },   // av
  };
  for (int bi = 0; bi < 9; ++bi) {
    BlockW w = mmW(bi);
    run_block(stream, w, mmcfg[bi].q, D0, mmcfg[bi].kv, D0, mmcfg[bi].mask,
              mmcfg[bi].out, D3, D0, DFF0, sc);   // write into concat buffer
  }

  // ---- fusion blocks (3 streams x L=2 layers, D=1536, dff=6144) ------------
  auto fuW = [&](int bi, int li) -> BlockW {
    BlockW w; const long long o = (long long)bi * 2 + li;
    w.ln_g = fu_p[0] + o * 4 * D3;  w.ln_b = fu_p[1] + o * 4 * D3;
    w.wq = fu_p[2] + o * D3 * D3;   w.bq = fu_p[3] + o * D3;
    w.wk = fu_p[4] + o * D3 * D3;   w.bk = fu_p[5] + o * D3;
    w.wv = fu_p[6] + o * D3 * D3;   w.bv = fu_p[7] + o * D3;
    w.wo = fu_p[8] + o * D3 * D3;   w.bo = fu_p[9] + o * D3;
    w.w1 = fu_p[10] + o * D3 * DFF3; w.b1 = fu_p[11] + o * DFF3;
    w.w2 = fu_p[12] + o * DFF3 * D3; w.b2 = fu_p[13] + o * D3;
    return w;
  };
  for (int li = 0; li < 2; ++li) {
    BlockW w = fuW(0, li);
    run_block(stream, w, lc, D3, lc, D3, lmask, lc, D3, D3, DFF3, sc);
  }
  for (int li = 0; li < 2; ++li) {
    BlockW w = fuW(1, li);
    run_block(stream, w, vc3, D3, vc3, D3, vmask, vc3, D3, D3, DFF3, sc);
  }
  for (int li = 0; li < 2; ++li) {
    BlockW w = fuW(2, li);
    run_block(stream, w, ac3, D3, ac3, D3, amask, ac3, D3, D3, DFF3, sc);
  }

  // ---- pooling + head ------------------------------------------------------
  pool_kernel<<<(B * D3 + 255) / 256, 256, 0, stream>>>(lc, ac3, vc3, B, S, D3, pooled);
  gemm_w(stream, pooled, 2 * D3, fc_w, D0, fc_b, nullptr, 0, 0,
         fcout, D0, nullptr, 0, B, D0, 2 * D3, false);
  layernorm_kernel<<<B, 256, 0, stream>>>(fcout, D0, ln_g, ln_b, fcln, D0, D0, EPS_LN, 1);
  gemm_w(stream, fcln, D0, cls_w, 2, cls_b, nullptr, 0, 0,
         outp, 2, nullptr, 0, B, 2, D0, false);
}
